// MambaBlock_38508676776018
// MI455X (gfx1250) — compile-verified, matrix operand order checked by reading, not run on hardware
//
#include <hip/hip_runtime.h>
#include <cstdint>
#include <cstddef>

// ---------------------------------------------------------------------------
// Problem constants (from reference)
// ---------------------------------------------------------------------------
#define DMODEL 1024
#define HSIZE  2048
#define BATCH  4
#define SEQ    2048
#define ROWS   (BATCH * SEQ)      // 8192
#define G3H    (3 * HSIZE)        // 6144
#define LN_EPS 1e-5f

// ---------------------------------------------------------------------------
// CDNA5 WMMA types
// ---------------------------------------------------------------------------
typedef __bf16          v16bf  __attribute__((ext_vector_type(16)));
typedef float           v8f    __attribute__((ext_vector_type(8)));
typedef unsigned short  ushort16 __attribute__((ext_vector_type(16)));

__device__ __forceinline__ unsigned short f2bf(float f) {
    unsigned u = __float_as_uint(f);
    unsigned rnd = ((u >> 16) & 1u) + 0x7FFFu;       // round-to-nearest-even
    return (unsigned short)((u + rnd) >> 16);
}

__device__ __forceinline__ v16bf load_bf16x16(const unsigned short* p) {
    ushort16 u = *reinterpret_cast<const ushort16*>(p);
    return __builtin_bit_cast(v16bf, u);
}

__device__ __forceinline__ float sigmoidf_(float x) {
    return 1.0f / (1.0f + __expf(-x));
}

#define WMMA_BF16(A_, B_, C_) \
    __builtin_amdgcn_wmma_f32_16x16x32_bf16(false, (A_), false, (B_), (short)0, (C_), false, false)

// ---------------------------------------------------------------------------
// f32 -> bf16 cast (weights / activations)
// ---------------------------------------------------------------------------
__global__ void cast_f32_to_bf16(const float* __restrict__ s,
                                 unsigned short* __restrict__ d, int n) {
    for (int i = blockIdx.x * blockDim.x + threadIdx.x; i < n;
         i += gridDim.x * blockDim.x)
        d[i] = f2bf(s[i]);
}

// ---------------------------------------------------------------------------
// LayerNorm over D=1024; one wave per row, 32 elements per lane.
// ---------------------------------------------------------------------------
__global__ __launch_bounds__(128)
void layernorm_kernel(const float* __restrict__ x,
                      const float* __restrict__ gamma,
                      const float* __restrict__ beta,
                      unsigned short* __restrict__ y_bf) {
    const int wave = threadIdx.x >> 5;
    const int lane = threadIdx.x & 31;
    const int row  = blockIdx.x * 4 + wave;
    if (row >= ROWS) return;
    const float* xr = x + (size_t)row * DMODEL;

    float v[32];
    float s = 0.f, sq = 0.f;
#pragma unroll
    for (int i = 0; i < 32; ++i) {
        float f = xr[lane + 32 * i];
        v[i] = f;
        s += f;
        sq += f * f;
    }
#pragma unroll
    for (int off = 16; off > 0; off >>= 1) {
        s  += __shfl_xor(s,  off);
        sq += __shfl_xor(sq, off);
    }
    const float mu  = s  * (1.0f / DMODEL);
    const float var = sq * (1.0f / DMODEL) - mu * mu;
    const float rs  = rsqrtf(var + LN_EPS);

    unsigned short* yr = y_bf + (size_t)row * DMODEL;
#pragma unroll
    for (int i = 0; i < 32; ++i) {
        int c = lane + 32 * i;
        float yv = (v[i] - mu) * rs * gamma[c] + beta[c];
        yr[c] = f2bf(yv);
    }
}

// ---------------------------------------------------------------------------
// Generic bf16 GEMM: out[M,N] = A[M,K] * W[N,K]^T (+bias[N]) (+resid[M,N])
// 16x64 tile per wave; software-pipelined: fragments for K-step k+32 are
// loaded into distinct registers while the WMMAs for K-step k execute.
// ---------------------------------------------------------------------------
__global__ __launch_bounds__(256)
void gemm_bf16_kernel(const unsigned short* __restrict__ A,
                      const unsigned short* __restrict__ W,
                      const float* __restrict__ bias,
                      const float* __restrict__ resid,
                      float* __restrict__ out,
                      int M, int N, int K) {
    const int lane   = threadIdx.x & 31;
    const int waveId = blockIdx.x * (blockDim.x >> 5) + (threadIdx.x >> 5);
    const int n64    = N >> 6;                       // N / 64
    const int tiles  = (M >> 4) * n64;
    if (waveId >= tiles) return;

    const int m0 = (waveId / n64) * 16;
    const int n0 = (waveId % n64) * 64;

    const int mrow = lane & 15;                      // A/B row within tile
    const int koff = (lane >> 4) << 4;               // K half select (0 / 16)

    const unsigned short* Arow = A + (size_t)(m0 + mrow) * K + koff;
    const unsigned short* W0   = W + (size_t)(n0 +  0 + mrow) * K + koff;
    const unsigned short* W1   = W + (size_t)(n0 + 16 + mrow) * K + koff;
    const unsigned short* W2   = W + (size_t)(n0 + 32 + mrow) * K + koff;
    const unsigned short* W3   = W + (size_t)(n0 + 48 + mrow) * K + koff;

    v8f acc0 = {}, acc1 = {}, acc2 = {}, acc3 = {};

    // prologue: fragments for k0 = 0
    v16bf a  = load_bf16x16(Arow);
    v16bf b0 = load_bf16x16(W0);
    v16bf b1 = load_bf16x16(W1);
    v16bf b2 = load_bf16x16(W2);
    v16bf b3 = load_bf16x16(W3);

    for (int k0 = 32; k0 < K; k0 += 32) {
        // issue next-step loads first: in flight under the WMMAs below
        v16bf an  = load_bf16x16(Arow + k0);
        v16bf c0  = load_bf16x16(W0 + k0);
        v16bf c1  = load_bf16x16(W1 + k0);
        v16bf c2  = load_bf16x16(W2 + k0);
        v16bf c3  = load_bf16x16(W3 + k0);

        acc0 = WMMA_BF16(a, b0, acc0);
        acc1 = WMMA_BF16(a, b1, acc1);
        acc2 = WMMA_BF16(a, b2, acc2);
        acc3 = WMMA_BF16(a, b3, acc3);

        a = an; b0 = c0; b1 = c1; b2 = c2; b3 = c3;
    }
    // epilogue K-step
    acc0 = WMMA_BF16(a, b0, acc0);
    acc1 = WMMA_BF16(a, b1, acc1);
    acc2 = WMMA_BF16(a, b2, acc2);
    acc3 = WMMA_BF16(a, b3, acc3);

    // C/D layout: vgpr r holds row m = r + (lane<16 ? 0 : 8), col n = lane&15
    const int mbase = m0 + ((lane >> 4) << 3);
    const int ncol  = lane & 15;
    v8f accs[4] = {acc0, acc1, acc2, acc3};
#pragma unroll
    for (int j = 0; j < 4; ++j) {
        const int n = n0 + j * 16 + ncol;
        const float bb = bias ? bias[n] : 0.0f;
#pragma unroll
        for (int r = 0; r < 8; ++r) {
            const size_t idx = (size_t)(mbase + r) * N + n;
            float v = accs[j][r] + bb;
            if (resid) v += resid[idx];
            out[idx] = v;
        }
    }
}

// ---------------------------------------------------------------------------
// Reset kernel: zero the global barrier counter and the h ping-pong buffers
// (graph replays must start from a clean state).
// ---------------------------------------------------------------------------
__global__ void reset_kernel(unsigned* __restrict__ bar,
                             unsigned short* __restrict__ hbuf) {
    if (blockIdx.x == 0 && threadIdx.x == 0) *bar = 0u;
    for (int i = threadIdx.x; i < 2 * BATCH * HSIZE; i += blockDim.x)
        hbuf[i] = 0;
}

// ---------------------------------------------------------------------------
// Persistent GRU scan. 32 WGs x 4 waves = 128 waves; wave w owns hidden units
// [16w, 16w+16) for the whole sequence: three gate rows via bf16 WMMA per
// step (batch rows 0..3 live in the M=16 tile), GRU update with previous h
// kept in registers, one software global barrier per step. bf16 w_hh (24 MB)
// stays resident in the 192 MB L2. Inner K-loop is software-pipelined.
// ---------------------------------------------------------------------------
__global__ __launch_bounds__(128)
void gru_scan_kernel(const float* __restrict__ xi,          // [ROWS, 3H] f32
                     const unsigned short* __restrict__ whh,// [3H, H] bf16
                     const float* __restrict__ b_hh,        // [3H] f32
                     unsigned short* __restrict__ hbuf,     // [2][B][H] bf16
                     unsigned short* __restrict__ hseq,     // [ROWS, H] bf16
                     unsigned* __restrict__ bar) {
    const int lane  = threadIdx.x & 31;
    const int waveG = blockIdx.x * 4 + (threadIdx.x >> 5);  // 0..127
    const int i0    = waveG * 16;

    const int mrow  = lane & 15;
    const int koff  = (lane >> 4) << 4;
    const bool aLive = (mrow < BATCH);

    const unsigned short* Wr = whh + (size_t)(i0 + mrow) * HSIZE + koff;
    const unsigned short* Wz = Wr + (size_t)HSIZE * HSIZE;
    const unsigned short* Wn = Wz + (size_t)HSIZE * HSIZE;

    float hp0 = 0.f, hp1 = 0.f, hp2 = 0.f, hp3 = 0.f;   // own previous h

    for (int t = 0; t < SEQ; ++t) {
        const unsigned short* hprev = hbuf + (size_t)(t & 1) * (BATCH * HSIZE);
        unsigned short* hnext = hbuf + (size_t)((t + 1) & 1) * (BATCH * HSIZE);
        const unsigned short* hrow = hprev + mrow * HSIZE + koff;

        v8f ar = {}, az = {}, an = {};

        // prologue fragments (k0 = 0)
        ushort16 au = {};
        if (aLive) au = *reinterpret_cast<const ushort16*>(hrow);
        v16bf a  = __builtin_bit_cast(v16bf, au);
        v16bf br = load_bf16x16(Wr);
        v16bf bz = load_bf16x16(Wz);
        v16bf bn = load_bf16x16(Wn);

        for (int k0 = 32; k0 < HSIZE; k0 += 32) {
            ushort16 aun = {};
            if (aLive) aun = *reinterpret_cast<const ushort16*>(hrow + k0);
            v16bf a2  = __builtin_bit_cast(v16bf, aun);
            v16bf cr  = load_bf16x16(Wr + k0);
            v16bf cz  = load_bf16x16(Wz + k0);
            v16bf cn  = load_bf16x16(Wn + k0);

            ar = WMMA_BF16(a, br, ar);
            az = WMMA_BF16(a, bz, az);
            an = WMMA_BF16(a, bn, an);

            a = a2; br = cr; bz = cz; bn = cn;
        }
        ar = WMMA_BF16(a, br, ar);
        az = WMMA_BF16(a, bz, az);
        an = WMMA_BF16(a, bn, an);

        // lanes 0..15 hold result rows m=0..7 in vgprs 0..7; batches are m=0..3
        if (lane < 16) {
            const int i = i0 + lane;
            const float bhr = b_hh[i];
            const float bhz = b_hh[HSIZE + i];
            const float bhn = b_hh[2 * HSIZE + i];
            float hps[4] = {hp0, hp1, hp2, hp3};
#pragma unroll
            for (int b = 0; b < BATCH; ++b) {
                const size_t base = (size_t)(b * SEQ + t) * G3H;
                float rr = sigmoidf_(xi[base + i]           + ar[b] + bhr);
                float zz = sigmoidf_(xi[base + HSIZE + i]   + az[b] + bhz);
                float nn = tanhf(xi[base + 2 * HSIZE + i] + rr * (an[b] + bhn));
                float hn = (1.0f - zz) * nn + zz * hps[b];
                hps[b] = hn;
                unsigned short hb = f2bf(hn);
                hnext[b * HSIZE + i] = hb;
                hseq[(size_t)(b * SEQ + t) * HSIZE + i] = hb;
            }
            hp0 = hps[0]; hp1 = hps[1]; hp2 = hps[2]; hp3 = hps[3];
        }

        // ---- one global barrier per step ----
        __threadfence();
        __syncthreads();
        if (threadIdx.x == 0) {
            __hip_atomic_fetch_add(bar, 1u, __ATOMIC_RELEASE,
                                   __HIP_MEMORY_SCOPE_AGENT);
            const unsigned target = (unsigned)(t + 1) * gridDim.x;
            while (__hip_atomic_load(bar, __ATOMIC_ACQUIRE,
                                     __HIP_MEMORY_SCOPE_AGENT) < target)
                __builtin_amdgcn_s_sleep(2);
        }
        __syncthreads();
    }
}

// ---------------------------------------------------------------------------
// Host-side launch
// ---------------------------------------------------------------------------
extern "C" void kernel_launch(void* const* d_in, const int* in_sizes, int n_in,
                              void* d_out, int out_size, void* d_ws, size_t ws_size,
                              hipStream_t stream) {
    (void)in_sizes; (void)n_in; (void)out_size; (void)ws_size;

    const float* x     = (const float*)d_in[0];
    const float* gamma = (const float*)d_in[1];
    const float* beta  = (const float*)d_in[2];
    const float* w_ih  = (const float*)d_in[3];
    const float* w_hh  = (const float*)d_in[4];
    const float* b_ih  = (const float*)d_in[5];
    const float* b_hh  = (const float*)d_in[6];
    const float* w_out = (const float*)d_in[7];
    const float* b_out = (const float*)d_in[8];
    float* out = (float*)d_out;

    // ---- workspace carve (256B aligned) ----
    char* ws = (char*)d_ws;
    size_t off = 0;
    auto carve = [&](size_t bytes) {
        void* p = ws + off;
        off += (bytes + 255) & ~(size_t)255;
        return p;
    };
    unsigned short* y_bf    = (unsigned short*)carve((size_t)ROWS * DMODEL * 2);
    unsigned short* wih_bf  = (unsigned short*)carve((size_t)G3H * DMODEL * 2);
    unsigned short* whh_bf  = (unsigned short*)carve((size_t)G3H * HSIZE * 2);
    unsigned short* wout_bf = (unsigned short*)carve((size_t)DMODEL * HSIZE * 2);
    unsigned short* hseq    = (unsigned short*)carve((size_t)ROWS * HSIZE * 2);
    unsigned short* hbuf    = (unsigned short*)carve((size_t)2 * BATCH * HSIZE * 2);
    unsigned*       bar     = (unsigned*)carve(256);
    float*          xi      = (float*)carve((size_t)ROWS * G3H * 4);

    // 1) weight casts to bf16
    cast_f32_to_bf16<<<4096, 256, 0, stream>>>(w_ih,  wih_bf,  G3H * DMODEL);
    cast_f32_to_bf16<<<4096, 256, 0, stream>>>(w_hh,  whh_bf,  G3H * HSIZE);
    cast_f32_to_bf16<<<4096, 256, 0, stream>>>(w_out, wout_bf, DMODEL * HSIZE);

    // 2) LayerNorm -> y (bf16)
    layernorm_kernel<<<ROWS / 4, 128, 0, stream>>>(x, gamma, beta, y_bf);

    // 3) GEMM1: xi = y * w_ih^T + b_ih   [8192 x 6144 x 1024]
    {
        int tiles = (ROWS / 16) * (G3H / 64);           // 49152
        gemm_bf16_kernel<<<tiles / 8, 256, 0, stream>>>(
            y_bf, wih_bf, b_ih, nullptr, xi, ROWS, G3H, DMODEL);
    }

    // 4) reset barrier + h ping-pong
    reset_kernel<<<1, 256, 0, stream>>>(bar, hbuf);

    // 5) persistent GRU scan (32 WGs x 4 waves = 128 waves, one per h-tile)
    gru_scan_kernel<<<32, 128, 0, stream>>>(xi, whh_bf, b_hh, hbuf, hseq, bar);

    // 6) GEMM3: out = x + hseq * w_out^T + b_out   [8192 x 1024 x 2048]
    {
        int tiles = (ROWS / 16) * (DMODEL / 64);        // 8192
        gemm_bf16_kernel<<<tiles / 8, 256, 0, stream>>>(
            hseq, wout_bf, b_out, x, out, ROWS, DMODEL, HSIZE);
    }
}
